// SwitchHeadCore_1666447311384
// MI455X (gfx1250) — matrix operand order repaired and checked by reading, not compiled, not run on hardware
//
#include <hip/hip_runtime.h>
#include <math.h>

// ---------------------------------------------------------------------------
// SwitchHead core for MI455X (gfx1250, wave32, WMMA).
// bf16 WMMA (f32 accumulate) for all GEMMs; fp32 softmax/gates/accums.
// Async global->LDS staging (ASYNCcnt) for attention tiles.
// ---------------------------------------------------------------------------

#define DEV __device__ __forceinline__

typedef __attribute__((ext_vector_type(16))) __bf16 v16bf;
typedef __attribute__((ext_vector_type(8)))  __bf16 v8bf;
typedef __attribute__((ext_vector_type(8)))  float  v8f;
typedef __attribute__((ext_vector_type(4)))  float  v4f;

namespace {
constexpr int Bb = 2, Ss = 2048, Dd = 1024, Hh = 8, Ee = 8, Pp = 128;
constexpr int NTOK = Bb * Ss;                       // 4096 tokens
constexpr float ATT_SCALE = 0.08838834764831845f;   // 1/sqrt(128)
}

// ---------------- WMMA fragment helpers (A/B k contiguous per lane half: k = 16h + s) ----------------
DEV v16bf cat8(v8bf lo, v8bf hi) {
  return __builtin_shufflevector(lo, hi, 0, 1, 2, 3, 4, 5, 6, 7,
                                 8, 9, 10, 11, 12, 13, 14, 15);
}

// A 16x32 bf16: lane<16 -> rows M=lane, K in {0..15};
//               lane>=16 -> same rows, K in {16..31}.
// Scaled per-row gather (fp32 or bf16 source).
template <typename T>
DEV v16bf a_frag(const T* a, int lda, float s) {
  const int lane = threadIdx.x & 31;
  const int r = lane & 15;
  const int kb = (lane < 16) ? 0 : 16;
  const T* row = a + (size_t)r * lda;
  v16bf f;
#pragma unroll
  for (int j = 0; j < 8; ++j) f[j] = (__bf16)((float)row[kb + j] * s);
#pragma unroll
  for (int j = 0; j < 8; ++j) f[8 + j] = (__bf16)((float)row[kb + 8 + j] * s);
  return f;
}

// Unscaled bf16 A fragment: two 16B vector loads per lane.
DEV v16bf a_frag_bf16(const __bf16* a, int lda) {
  const int lane = threadIdx.x & 31;
  const int r = lane & 15;
  const int kb = (lane < 16) ? 0 : 16;
  const __bf16* row = a + (size_t)r * lda;
  const v8bf lo = *(const v8bf*)(row + kb);
  const v8bf hi = *(const v8bf*)(row + kb + 8);
  return cat8(lo, hi);
}

// B 32x16, source stored K-major (row index = K): strided gather.
template <typename T>
DEV v16bf b_frag_k(const T* b, int ldb) {
  const int lane = threadIdx.x & 31;
  const int n = lane & 15;
  const int kb = (lane < 16) ? 0 : 16;
  v16bf f;
#pragma unroll
  for (int j = 0; j < 16; ++j) f[j] = (__bf16)(float)b[(size_t)(kb + j) * ldb + n];
  return f;
}

// B 32x16, source stored N-major (row = N, cols = K): contiguous vector loads.
DEV v16bf b_frag_n_bf16(const __bf16* b, int ldn) {
  const int lane = threadIdx.x & 31;
  const int n = lane & 15;
  const int kb = (lane < 16) ? 0 : 16;
  const __bf16* row = b + (size_t)n * ldn + kb;
  const v8bf lo = *(const v8bf*)(row);
  const v8bf hi = *(const v8bf*)(row + 8);
  return cat8(lo, hi);
}

DEV v16bf b_frag_n_f32(const float* b, int ldn) {
  const int lane = threadIdx.x & 31;
  const int n = lane & 15;
  const int kb = (lane < 16) ? 0 : 16;
  const float* row = b + (size_t)n * ldn + kb;
  const v4f q0 = *(const v4f*)(row);
  const v4f q1 = *(const v4f*)(row + 4);
  const v4f q2 = *(const v4f*)(row + 8);
  const v4f q3 = *(const v4f*)(row + 12);
  v16bf f;
#pragma unroll
  for (int j = 0; j < 4; ++j) {
    f[j]      = (__bf16)q0[j];
    f[4 + j]  = (__bf16)q1[j];
    f[8 + j]  = (__bf16)q2[j];
    f[12 + j] = (__bf16)q3[j];
  }
  return f;
}

DEV v8f wmma_bf16(v16bf a, v16bf b, v8f c) {
  // 8 args: (neg_a, A, neg_b, B, c_mod, C, reuse_a, reuse_b)
  return __builtin_amdgcn_wmma_f32_16x16x32_bf16(false, a, false, b, (short)0, c,
                                                 false, false);
}

// C/D f32 16x16: lane<16 -> N=lane, M=r; lane>=16 -> N=lane-16, M=8+r.
DEV void c_coords(int& n, int& mo) {
  const int lane = threadIdx.x & 31;
  n = lane & 15;
  mo = (lane < 16) ? 0 : 8;
}

// ------------- CDNA5 async global->LDS copy (ASYNCcnt tracked) --------------
// GV mode: vdst = LDS byte offset (low 32 bits of generic shared pointer),
// vaddr = 64-bit global address.  ISA 10 / 15.18.3 opcode 98.
DEV void async_copy_b128(uint32_t lds_off, const void* gptr) {
  asm volatile("global_load_async_to_lds_b128 %0, %1, off"
               :: "v"(lds_off), "v"((uint64_t)(uintptr_t)gptr)
               : "memory");
}
DEV void async_wait0() { asm volatile("s_wait_asynccnt 0" ::: "memory"); }

// ---------------------------------------------------------------------------
// K1: routing gates. grid = NTOK blocks x 64 threads.
// ---------------------------------------------------------------------------
__global__ void gates_kernel(const float* __restrict__ x,
                             const float* __restrict__ sel_v_w,
                             const float* __restrict__ sel_o_w,
                             const float* __restrict__ route_scale,
                             float* __restrict__ gate_v,
                             float* __restrict__ gate_o) {
  __shared__ float xs[Dd];
  __shared__ float pv[64];
  __shared__ float po[64];
  const int t = blockIdx.x;
  const float* xr = x + (size_t)t * Dd;
  for (int i = threadIdx.x; i < Dd; i += 64) xs[i] = xr[i];
  __syncthreads();

  const int he = threadIdx.x;  // 0..63 == h*8+e
  const float* wv = sel_v_w + (size_t)he * Dd;
  const float* wo = sel_o_w + (size_t)he * Dd;
  float sv = 0.f, so = 0.f;
  for (int d = 0; d < Dd; ++d) {
    const float xv = xs[d];
    sv = fmaf(xv, wv[d], sv);
    so = fmaf(xv, wo[d], so);
  }
  pv[he] = 1.f / (1.f + __expf(-sv));
  po[he] = 1.f / (1.f + __expf(-so));
  __syncthreads();

  if (threadIdx.x < 16) {
    const int h = threadIdx.x & 7;
    float* p = (threadIdx.x < 8) ? (pv + h * 8) : (po + h * 8);
    float* g = ((threadIdx.x < 8) ? gate_v : gate_o) + ((size_t)t * Hh + h) * Ee;
    int i1 = 0;
    for (int e = 1; e < 8; ++e) if (p[e] > p[i1]) i1 = e;
    int i2 = -1;
    for (int e = 0; e < 8; ++e) if (e != i1 && (i2 < 0 || p[e] > p[i2])) i2 = e;
    const float s = fmaxf(p[i1] + p[i2], 1e-9f);
    const float rs = route_scale[0];
#pragma unroll
    for (int e = 0; e < 8; ++e) g[e] = 0.f;
    g[i1] = p[i1] / s * rs;
    g[i2] = p[i2] / s * rs;
  }
}

// ---------------------------------------------------------------------------
// K2: q/k projection GEMM -> bf16 [B,H,S,P].
// grid = (NTOK/16, (H*P)/64, 2[q|k]); block = 128 (4 waves, one 16x16 C each)
// ---------------------------------------------------------------------------
__global__ void qk_proj_kernel(const float* __restrict__ x,
                               const float* __restrict__ q_w,
                               const float* __restrict__ k_w,
                               __bf16* __restrict__ qg,
                               __bf16* __restrict__ kg) {
  const float* w = blockIdx.z ? k_w : q_w;
  __bf16* out = blockIdx.z ? kg : qg;
  const int t0 = blockIdx.x * 16;
  const int n0 = blockIdx.y * 64;
  const int wv = threadIdx.x >> 5;
  const int nw = n0 + wv * 16;

  __shared__ __bf16 xa[16 * 32];  // x tile pre-converted to bf16
  v8f acc = {};
  for (int d0 = 0; d0 < Dd; d0 += 32) {
    for (int i = threadIdx.x; i < 16 * 32; i += 128) {
      const int r = i >> 5, c = i & 31;
      xa[i] = (__bf16)x[(size_t)(t0 + r) * Dd + d0 + c];
    }
    __builtin_prefetch(w + (size_t)nw * Dd + d0 + 32, 0, 0);
    __syncthreads();
    const v16bf a = a_frag_bf16(xa, 32);
    const v16bf b = b_frag_n_f32(w + (size_t)nw * Dd + d0, Dd);
    acc = wmma_bf16(a, b, acc);
    __syncthreads();
  }

  int n, mo;
  c_coords(n, mo);
  const int col = nw + n;          // h*P + p
  const int h = col >> 7, p = col & (Pp - 1);
#pragma unroll
  for (int r = 0; r < 8; ++r) {
    const int t = t0 + mo + r;
    const int bI = t >> 11, s = t & (Ss - 1);
    out[(((size_t)bI * Hh + h) * Ss + s) * Pp + p] = (__bf16)acc[r];
  }
}

// ---------------------------------------------------------------------------
// K3: gated expert V projection. v = sum_e diag(g_e) * (X * W_he).
// Gate folded into A-fragment row scaling -> single accumulator over (e,d).
// grid = (NTOK/16, H, P/64); block = 128.
// ---------------------------------------------------------------------------
__global__ void v_proj_kernel(const float* __restrict__ x,
                              const float* __restrict__ v_w,
                              const float* __restrict__ gate_v,
                              __bf16* __restrict__ vg) {
  const int t0 = blockIdx.x * 16;
  const int h = blockIdx.y;
  const int p0 = blockIdx.z * 64;
  const int wv = threadIdx.x >> 5;
  const int pw = p0 + wv * 16;
  const int lane = threadIdx.x & 31;
  const int ar = lane & 15;

  __shared__ float xa[16 * 32];
  __shared__ float g[16][8];
  __shared__ float emax[8];

  {
    const int r = threadIdx.x >> 3, e = threadIdx.x & 7;
    g[r][e] = gate_v[((size_t)(t0 + r) * Hh + h) * Ee + e];
  }
  __syncthreads();
  if (threadIdx.x < 8) {
    float m = 0.f;
    for (int r = 0; r < 16; ++r) m = fmaxf(m, fabsf(g[r][threadIdx.x]));
    emax[threadIdx.x] = m;
  }
  __syncthreads();

  v8f acc = {};
  for (int d0 = 0; d0 < Dd; d0 += 32) {
    for (int i = threadIdx.x; i < 16 * 32; i += 128) {
      const int r = i >> 5, c = i & 31;
      xa[i] = x[(size_t)(t0 + r) * Dd + d0 + c];
    }
    __syncthreads();
    for (int e = 0; e < Ee; ++e) {
      if (emax[e] == 0.f) continue;  // block-uniform skip: EXEC stays full
      const v16bf a = a_frag(xa, 32, g[ar][e]);
      const float* bsrc = v_w + (((size_t)h * Ee + e) * Dd + d0) * Pp + pw;
      const v16bf b = b_frag_k(bsrc, Pp);
      acc = wmma_bf16(a, b, acc);
    }
    __syncthreads();
  }

  int n, mo;
  c_coords(n, mo);
#pragma unroll
  for (int r = 0; r < 8; ++r) {
    const int t = t0 + mo + r;
    const int bI = t >> 11, s = t & (Ss - 1);
    vg[(((size_t)bI * Hh + h) * Ss + s) * Pp + pw + n] = (__bf16)acc[r];
  }
}

// ---------------------------------------------------------------------------
// K4: attention for one (b,h,32-query tile). Whole 32x2048 score row block
// lives in LDS (256KB of the 320KB WGP LDS). grid=(S/32,H,B); block=256.
// dyn smem: sc[32][2048] f32 | qa[32][128] bf16 | kv[64][128] bf16
// Q/K/V tiles staged with GLOBAL_LOAD_ASYNC_TO_LDS_B128.
// ---------------------------------------------------------------------------
__global__ void attn_kernel(const __bf16* __restrict__ qg,
                            const __bf16* __restrict__ kg,
                            const __bf16* __restrict__ vg,
                            __bf16* __restrict__ ctx) {
  extern __shared__ char smem[];
  float* sc = (float*)smem;                                     // 262144 B
  __bf16* qa = (__bf16*)(smem + 32 * Ss * sizeof(float));       // 8192 B
  __bf16* kv = qa + 32 * Pp;                                    // 16384 B
  __shared__ float rsum[32];

  const int q0 = blockIdx.x * 32;
  const int h = blockIdx.y;
  const int b = blockIdx.z;
  const size_t bh = ((size_t)b * Hh + h) * Ss;  // base row (in S units)
  const int tid = threadIdx.x;
  const int wv = tid >> 5;
  const int lane = tid & 31;
  const uint32_t qa_lds = (uint32_t)(uintptr_t)(void*)qa;
  const uint32_t kv_lds = (uint32_t)(uintptr_t)(void*)kv;

  // stage Q tile (32x128 bf16 = 8KB) asynchronously
  {
    const char* src = (const char*)(qg + (bh + q0) * Pp);
    for (int i = tid; i < 512; i += 256)
      async_copy_b128(qa_lds + i * 16, src + (size_t)i * 16);
    async_wait0();
  }
  __syncthreads();

  // ---- Phase 1: S = (Q K^T) * scale, into LDS ----
  {
    const int mrow = (wv >> 2) * 16;
    const int ncol = (wv & 3) * 16;
    for (int kt = 0; kt < Ss; kt += 64) {
      const char* src = (const char*)(kg + (bh + kt) * Pp);  // 64x128 bf16
      for (int i = tid; i < 1024; i += 256)
        async_copy_b128(kv_lds + i * 16, src + (size_t)i * 16);
      async_wait0();
      __syncthreads();
      v8f acc = {};
#pragma unroll
      for (int pp = 0; pp < Pp; pp += 32) {
        const v16bf a = a_frag_bf16(qa + mrow * Pp + pp, Pp);
        const v16bf bf = b_frag_n_bf16(kv + ncol * Pp + pp, Pp);
        acc = wmma_bf16(a, bf, acc);
      }
      int n, mo;
      c_coords(n, mo);
#pragma unroll
      for (int r = 0; r < 8; ++r)
        sc[(size_t)(mrow + mo + r) * Ss + kt + ncol + n] = acc[r] * ATT_SCALE;
      __syncthreads();
    }
  }

  // ---- Phase 2: fp32 softmax, one wave per 4 rows ----
  for (int r = wv * 4; r < wv * 4 + 4; ++r) {
    float* row = sc + (size_t)r * Ss;
    float m = -3.0e38f;
    for (int i = lane; i < Ss; i += 32) m = fmaxf(m, row[i]);
#pragma unroll
    for (int off = 16; off > 0; off >>= 1) m = fmaxf(m, __shfl_xor(m, off, 32));
    float s = 0.f;
    for (int i = lane; i < Ss; i += 32) {
      const float e = __expf(row[i] - m);
      row[i] = e;
      s += e;
    }
#pragma unroll
    for (int off = 16; off > 0; off >>= 1) s += __shfl_xor(s, off, 32);
    if (lane == 0) rsum[r] = 1.0f / s;
  }
  __syncthreads();

  // ---- Phase 3: ctx = P V (1/rowsum folded into A gather) ----
  v8f acc0 = {}, acc1 = {};
  const int ct0 = wv * 2, ct1 = wv * 2 + 1;
  const int m0r = (ct0 >> 3) * 16, n0c = (ct0 & 7) * 16;
  const int m1r = (ct1 >> 3) * 16, n1c = (ct1 & 7) * 16;
  for (int kt = 0; kt < Ss; kt += 32) {
    const char* src = (const char*)(vg + (bh + kt) * Pp);  // 32x128 bf16
    for (int i = tid; i < 512; i += 256)
      async_copy_b128(kv_lds + i * 16, src + (size_t)i * 16);
    async_wait0();
    __syncthreads();
    {
      const v16bf a = a_frag(sc + (size_t)m0r * Ss + kt, Ss, rsum[m0r + (lane & 15)]);
      const v16bf bf = b_frag_k(kv + n0c, Pp);
      acc0 = wmma_bf16(a, bf, acc0);
    }
    {
      const v16bf a = a_frag(sc + (size_t)m1r * Ss + kt, Ss, rsum[m1r + (lane & 15)]);
      const v16bf bf = b_frag_k(kv + n1c, Pp);
      acc1 = wmma_bf16(a, bf, acc1);
    }
    __syncthreads();
  }

  int n, mo;
  c_coords(n, mo);
#pragma unroll
  for (int r = 0; r < 8; ++r) {
    int tok = q0 + m0r + mo + r;
    ctx[(((size_t)b * Ss + tok) * Hh + h) * Pp + n0c + n] = (__bf16)acc0[r];
    tok = q0 + m1r + mo + r;
    ctx[(((size_t)b * Ss + tok) * Hh + h) * Pp + n1c + n] = (__bf16)acc1[r];
  }
}

// ---------------------------------------------------------------------------
// K5: gated expert O projection. out[t,d] = sum_{h,e} g_o * (ctx_h W_heT).
// grid = (NTOK/16, D/64); block = 128.
// ---------------------------------------------------------------------------
__global__ void o_proj_kernel(const __bf16* __restrict__ ctx,
                              const float* __restrict__ o_w,
                              const float* __restrict__ gate_o,
                              float* __restrict__ out) {
  const int t0 = blockIdx.x * 16;
  const int d0 = blockIdx.y * 64;
  const int wv = threadIdx.x >> 5;
  const int dn = d0 + wv * 16;
  const int lane = threadIdx.x & 31;
  const int ar = lane & 15;

  __shared__ __bf16 ca[16 * Pp];  // ctx tile for current head
  __shared__ float g[16][8];
  __shared__ float emax[8];

  v8f acc = {};
  for (int h = 0; h < Hh; ++h) {
    {
      const int r = threadIdx.x >> 3, e = threadIdx.x & 7;
      g[r][e] = gate_o[((size_t)(t0 + r) * Hh + h) * Ee + e];
    }
    __syncthreads();
    if (threadIdx.x < 8) {
      float m = 0.f;
      for (int r = 0; r < 16; ++r) m = fmaxf(m, fabsf(g[r][threadIdx.x]));
      emax[threadIdx.x] = m;
    }
    // stage ctx[t0..t0+15][h][0..127] (bf16) as dwords
    {
      const uint32_t* srcb = (const uint32_t*)ctx;
      uint32_t* dst = (uint32_t*)ca;
      for (int i = threadIdx.x; i < 16 * 64; i += 128) {
        const int r = i >> 6, c = i & 63;
        dst[i] = srcb[((size_t)(t0 + r) * Hh + h) * 64 + c];
      }
    }
    __syncthreads();

    for (int e = 0; e < Ee; ++e) {
      if (emax[e] == 0.f) continue;  // block-uniform
      const float gr = g[ar][e];
      const float* bbase = o_w + ((size_t)h * Ee + e) * Pp * Dd + dn;
#pragma unroll
      for (int pp = 0; pp < Pp; pp += 32) {
        __builtin_prefetch(bbase + (size_t)(pp + 32) * Dd, 0, 0);
        const v16bf a = a_frag(ca + pp, Pp, gr);
        const v16bf b = b_frag_k(bbase + (size_t)pp * Dd, Dd);
        acc = wmma_bf16(a, b, acc);
      }
    }
    __syncthreads();
  }

  int n, mo;
  c_coords(n, mo);
#pragma unroll
  for (int r = 0; r < 8; ++r)
    out[(size_t)(t0 + mo + r) * Dd + dn + n] = acc[r];
}

// ---------------------------------------------------------------------------
// kernel_launch
// inputs: x, q_w, k_w, v_w, o_w, sel_v_w, sel_o_w, route_scale
// ---------------------------------------------------------------------------
extern "C" void kernel_launch(void* const* d_in, const int* in_sizes, int n_in,
                              void* d_out, int out_size, void* d_ws, size_t ws_size,
                              hipStream_t stream) {
  (void)in_sizes; (void)n_in; (void)out_size; (void)ws_size;
  const float* x    = (const float*)d_in[0];
  const float* q_w  = (const float*)d_in[1];
  const float* k_w  = (const float*)d_in[2];
  const float* v_w  = (const float*)d_in[3];
  const float* o_w  = (const float*)d_in[4];
  const float* sv_w = (const float*)d_in[5];
  const float* so_w = (const float*)d_in[6];
  const float* rs   = (const float*)d_in[7];
  float* out = (float*)d_out;

  // workspace carve (all sizes 256B-aligned by construction); total ~34 MB
  char* w = (char*)d_ws;
  float*  gate_v = (float*)w;  w += (size_t)NTOK * Hh * Ee * sizeof(float);   // 1 MB
  float*  gate_o = (float*)w;  w += (size_t)NTOK * Hh * Ee * sizeof(float);   // 1 MB
  __bf16* qg  = (__bf16*)w;    w += (size_t)NTOK * Hh * Pp * sizeof(__bf16);  // 8 MB
  __bf16* kg  = (__bf16*)w;    w += (size_t)NTOK * Hh * Pp * sizeof(__bf16);  // 8 MB
  __bf16* vg  = (__bf16*)w;    w += (size_t)NTOK * Hh * Pp * sizeof(__bf16);  // 8 MB
  __bf16* ctx = (__bf16*)w;    w += (size_t)NTOK * Hh * Pp * sizeof(__bf16);  // 8 MB

  constexpr int ATTN_SMEM = 32 * Ss * (int)sizeof(float)   // score block
                          + 32 * Pp * (int)sizeof(__bf16)  // Q tile
                          + 64 * Pp * (int)sizeof(__bf16); // K/V staging
  (void)hipFuncSetAttribute((const void*)attn_kernel,
                            hipFuncAttributeMaxDynamicSharedMemorySize, ATTN_SMEM);

  gates_kernel<<<NTOK, 64, 0, stream>>>(x, sv_w, so_w, rs, gate_v, gate_o);
  qk_proj_kernel<<<dim3(NTOK / 16, (Hh * Pp) / 64, 2), 128, 0, stream>>>(
      x, q_w, k_w, qg, kg);
  v_proj_kernel<<<dim3(NTOK / 16, Hh, Pp / 64), 128, 0, stream>>>(
      x, v_w, gate_v, vg);
  attn_kernel<<<dim3(Ss / 32, Hh, Bb), 256, ATTN_SMEM, stream>>>(qg, kg, vg, ctx);
  o_proj_kernel<<<dim3(NTOK / 16, Dd / 64), 128, 0, stream>>>(ctx, o_w, gate_o, out);
}
